// DVGTformer_model_83150566851151
// MI455X (gfx1250) — compile-verified, matrix operand order checked by reading, not run on hardware
//
#include <hip/hip_runtime.h>
#include <hip/hip_bf16.h>
#include <math.h>

// ---------------------------------------------------------------------------
// Types for CDNA5 WMMA
// ---------------------------------------------------------------------------
typedef __bf16 v16bf __attribute__((ext_vector_type(16)));
typedef float  v8f   __attribute__((ext_vector_type(8)));
typedef unsigned short v8us  __attribute__((ext_vector_type(8)));   // 16 B
typedef unsigned short v16us __attribute__((ext_vector_type(16)));  // 32 B

__device__ __forceinline__ unsigned short f2bf(float f) {
    unsigned int u = __builtin_bit_cast(unsigned int, f);
    unsigned int r = u + 0x7FFFu + ((u >> 16) & 1u);   // round-to-nearest-even
    return (unsigned short)(r >> 16);
}
__device__ __forceinline__ float bf2f(unsigned short h) {
    unsigned int u = ((unsigned int)h) << 16;
    return __builtin_bit_cast(float, u);
}
__device__ __forceinline__ float gelu_f(float x) {
    return 0.5f * x * (1.f + erff(x * 0.70710678118654752f));
}

// block reductions (blockDim.x threads, power of two)
__device__ __forceinline__ float bsum(float* red, float v) {
    int t = threadIdx.x;
    red[t] = v; __syncthreads();
    for (int s = (int)blockDim.x >> 1; s > 0; s >>= 1) {
        if (t < s) red[t] += red[t + s];
        __syncthreads();
    }
    float r = red[0]; __syncthreads(); return r;
}
__device__ __forceinline__ float bmax(float* red, float v) {
    int t = threadIdx.x;
    red[t] = v; __syncthreads();
    for (int s = (int)blockDim.x >> 1; s > 0; s >>= 1) {
        if (t < s) red[t] = fmaxf(red[t], red[t + s]);
        __syncthreads();
    }
    float r = red[0]; __syncthreads(); return r;
}

// ---------------------------------------------------------------------------
// Batched BF16 WMMA GEMM, all-vectorized, branch-free inner loop.
//   C[z] = alpha * A[z] @ B[z]^T(+bias) (+GELU)
// A: (M, Kp) bf16 row-major, lda multiple of 32, zero-padded cols K..Kp.
// B: stored (N, Kp) row-major (pre-transposed), ldb multiple of 32, zero-pad.
// Kp: multiple of 32 (padded K); no ragged-K handling needed.
// Out-of-range rows read from Zg, a zero-filled guard buffer (>= Kp + 128),
// so the K loop has no predication at all.
// Outputs (exactly one used): C f32 (ldc) or Cbf bf16 (ldcb).
// transC: write Cbf transposed as (N, M); rows [M, mzero) zeroed (V^T path).
// Block tile 64x128x32; 8 waves as 2x4, each wave owns 32x32 (4 WMMA accs).
// LDS tiles are fragment-major: each lane's 16 operand values are 32
// contiguous bytes -> ds_load_b128 pairs; staging is global_load_b128 +
// ds_store_b128 only.
// ---------------------------------------------------------------------------
__launch_bounds__(256)
__global__ void gemm_bf16_wmma(const unsigned short* __restrict__ A,
                               const unsigned short* __restrict__ B,
                               const unsigned short* __restrict__ Zg,
                               const float* __restrict__ bias,
                               float* __restrict__ C,
                               unsigned short* __restrict__ Cbf,
                               int M, int N, int Kp,
                               int lda, int ldb, int ldc, int ldcb,
                               long long sA, long long sB, long long sC,
                               int doGelu, int transC, int mzero, float alpha)
{
    __shared__ __align__(32) unsigned short sAf[4][32][16];  // [mfrag][lane][slot]
    __shared__ __align__(32) unsigned short sBf[8][32][16];  // [nfrag][lane][slot]

    const int zb = blockIdx.z;
    const unsigned short* Ab = A + (long long)zb * sA;
    const unsigned short* Bb = B + (long long)zb * sB;

    const int bm   = blockIdx.y * 64;
    const int bn   = blockIdx.x * 128;
    const int tid  = threadIdx.x;
    const int lane = tid & 31;
    const int wave = tid >> 5;
    const int wrow = wave >> 2;   // 0..1 -> 32 rows
    const int wcol = wave & 3;    // 0..3 -> 32 cols

    // --- staging task decode (8 contiguous bf16 per task) ---
    // A: 256 tasks = 4 frags x 32 lanes x 2 slot-groups
    const int a_frag = tid >> 6;
    const int a_lane = (tid >> 1) & 31;
    const int a_grp  = tid & 1;
    const int a_row  = bm + (a_lane & 15) + a_frag * 16;
    const int a_kofs = a_grp * 16 + (a_lane >> 4) * 8;   // slot-group -> k offset
    const unsigned short* a_src =
        (a_row < M) ? (Ab + (long long)a_row * lda + a_kofs) : (Zg + a_kofs);

    // B: 512 tasks = 8 frags x 32 lanes x 2 slot-groups (2 per thread)
    int b_frag[2], b_lane[2], b_grp[2];
    const unsigned short* b_src[2];
#pragma unroll
    for (int p = 0; p < 2; ++p) {
        int t = tid + p * 256;
        b_frag[p] = t >> 6;
        b_lane[p] = (t >> 1) & 31;
        b_grp[p]  = t & 1;
        int row  = bn + (b_lane[p] & 15) + b_frag[p] * 16;
        int kofs = b_grp[p] * 16 + (b_lane[p] >> 4) * 8;
        b_src[p] = (row < N) ? (Bb + (long long)row * ldb + kofs) : (Zg + kofs);
    }

    const v8f vz = {0.f, 0.f, 0.f, 0.f, 0.f, 0.f, 0.f, 0.f};
    v8f acc00 = vz, acc01 = vz, acc10 = vz, acc11 = vz;

    for (int k0 = 0; k0 < Kp; k0 += 32) {
        v8us av  = *(const v8us*)(a_src + k0);
        v8us bv0 = *(const v8us*)(b_src[0] + k0);
        v8us bv1 = *(const v8us*)(b_src[1] + k0);
        // CDNA5 speculative prefetch (global_prefetch_b8), 2 tiles ahead;
        // tail overshoot lands in guard/adjacent ws and is dropped silently.
        __builtin_prefetch(a_src + k0 + 64, 0, 1);
        __builtin_prefetch(b_src[0] + k0 + 64, 0, 1);
        __syncthreads();   // previous iteration's fragment reads done
        *(v8us*)&sAf[a_frag][a_lane][a_grp * 8]          = av;
        *(v8us*)&sBf[b_frag[0]][b_lane[0]][b_grp[0] * 8] = bv0;
        *(v8us*)&sBf[b_frag[1]][b_lane[1]][b_grp[1] * 8] = bv1;
        __syncthreads();

        v16bf a0 = __builtin_bit_cast(v16bf, *(const v16us*)&sAf[wrow * 2 + 0][lane][0]);
        v16bf a1 = __builtin_bit_cast(v16bf, *(const v16us*)&sAf[wrow * 2 + 1][lane][0]);
        v16bf b0 = __builtin_bit_cast(v16bf, *(const v16us*)&sBf[wcol * 2 + 0][lane][0]);
        v16bf b1 = __builtin_bit_cast(v16bf, *(const v16us*)&sBf[wcol * 2 + 1][lane][0]);
        acc00 = __builtin_amdgcn_wmma_f32_16x16x32_bf16(false, a0, false, b0, (short)0, acc00, false, false);
        acc01 = __builtin_amdgcn_wmma_f32_16x16x32_bf16(false, a0, false, b1, (short)0, acc01, false, false);
        acc10 = __builtin_amdgcn_wmma_f32_16x16x32_bf16(false, a1, false, b0, (short)0, acc10, false, false);
        acc11 = __builtin_amdgcn_wmma_f32_16x16x32_bf16(false, a1, false, b1, (short)0, acc11, false, false);
    }

    // --- epilogue: C/D layout -> n = lane&15 ; VGPR r -> m = r + 8*(lane>>4)
    v8f accs[2][2] = {{acc00, acc01}, {acc10, acc11}};
    const int nl = lane & 15;
    const int mh = (lane >> 4) << 3;
#pragma unroll
    for (int im = 0; im < 2; ++im) {
#pragma unroll
        for (int jn = 0; jn < 2; ++jn) {
            int gn  = bn + wcol * 32 + jn * 16 + nl;
            int gmb = bm + wrow * 32 + im * 16 + mh;
            if (gn >= N) continue;
            float bvv = bias ? bias[gn] : 0.f;
#pragma unroll
            for (int r = 0; r < 8; ++r) {
                int gm = gmb + r;
                float v = accs[im][jn][r] * alpha + bvv;
                if (doGelu) v = gelu_f(v);
                if (!transC) {
                    if (gm < M) {
                        if (C)   C[(long long)zb * sC + (long long)gm * ldc + gn] = v;
                        if (Cbf) Cbf[(long long)zb * sC + (long long)gm * ldcb + gn] = f2bf(v);
                    }
                } else {
                    long long off = (long long)zb * sC + (long long)gn * ldcb + gm;
                    if (gm < M)          Cbf[off] = f2bf(v);
                    else if (gm < mzero) Cbf[off] = 0;   // zero-pad K for next GEMM
                }
            }
        }
    }
}

// ---------------------------------------------------------------------------
// Casts / fills
// ---------------------------------------------------------------------------
__global__ void zero_us_kernel(unsigned short* __restrict__ out, long long n)
{
    long long i = (long long)blockIdx.x * blockDim.x + threadIdx.x;
    long long step = (long long)gridDim.x * blockDim.x;
    for (; i < n; i += step) out[i] = 0;
}

__global__ void cast_bf16_kernel(const float* __restrict__ in,
                                 unsigned short* __restrict__ out, long long n)
{
    long long i = (long long)blockIdx.x * blockDim.x + threadIdx.x;
    long long step = (long long)gridDim.x * blockDim.x;
    for (; i < n; i += step) out[i] = f2bf(in[i]);
}

// in: G blocks of (Rin, C) f32 -> out: G blocks of (C, Rpad) bf16 (B^T layout,
// zero rows Rin..Rpad).  out[(g*C + c)*Rpad + r]
__global__ void cast_bT_kernel(const float* __restrict__ in,
                               unsigned short* __restrict__ out,
                               int G, int Rin, int Rpad, int C)
{
    long long total = (long long)G * C * Rpad;
    long long i = (long long)blockIdx.x * blockDim.x + threadIdx.x;
    long long step = (long long)gridDim.x * blockDim.x;
    for (; i < total; i += step) {
        int r = (int)(i % Rpad);
        long long q = i / Rpad;
        int c = (int)(q % C);
        int g = (int)(q / C);
        float v = (r < Rin) ? in[((long long)g * Rin + r) * C + c] : 0.f;
        out[i] = f2bf(v);
    }
}

// Wout1 (513*257, 100) -> (100, 513*288) with interleaved row padding so the
// padded flattened activation (t*288+n) lines up.  out[(c*513+g)*288+r]
__global__ void cast_wout1_kernel(const float* __restrict__ in,
                                  unsigned short* __restrict__ out)
{
    long long total = 100ll * 513 * 288;
    long long i = (long long)blockIdx.x * blockDim.x + threadIdx.x;
    long long step = (long long)gridDim.x * blockDim.x;
    for (; i < total; i += step) {
        int r = (int)(i % 288);
        long long q = i / 288;
        int g = (int)(q % 513);
        int c = (int)(q / 513);
        float v = (r < 257) ? in[((long long)g * 257 + r) * 100 + c] : 0.f;
        out[i] = f2bf(v);
    }
}

// ---------------------------------------------------------------------------
// (B, M, ldin) -> (B, N, ldout) transpose, zero-filling pad cols [M, ldout).
// grid: x=ceil(N/32), y=ldout/32, z=B
// ---------------------------------------------------------------------------
__global__ void transpose_kernel(const float* __restrict__ in,
                                 float* __restrict__ out,
                                 unsigned short* __restrict__ outbf,
                                 int M, int N, int ldin, int ldout)
{
    __shared__ float tile[32][33];
    int b = blockIdx.z;
    const float* pin = in + (long long)b * M * ldin;
    float* pout = out + (long long)b * N * ldout;
    unsigned short* pobf = outbf + (long long)b * N * ldout;
    int m0 = blockIdx.y * 32, n0 = blockIdx.x * 32;
    int tx = threadIdx.x & 31, ty = threadIdx.x >> 5;   // 32x8
    for (int i = ty; i < 32; i += 8) {
        int m = m0 + i, n = n0 + tx;
        tile[i][tx] = (m < M && n < N) ? pin[(long long)m * ldin + n] : 0.f;
    }
    __syncthreads();
    for (int i = ty; i < 32; i += 8) {
        int n = n0 + i, m = m0 + tx;
        if (n < N && m < ldout) {
            float v = tile[tx][i];   // zero when m >= M
            pout[(long long)n * ldout + m] = v;
            pobf[(long long)n * ldout + m] = f2bf(v);
        }
    }
}

// assemble (16,513,288pad) from (16,512,256) + t_v row + x_v col; zero pads
__global__ void build_xc_kernel(const float* __restrict__ X2,
                                const float* __restrict__ tv,
                                const float* __restrict__ xv,
                                float* __restrict__ Xf,
                                unsigned short* __restrict__ Xbf)
{
    long long total = 16ll * 513 * 288;
    long long i = (long long)blockIdx.x * blockDim.x + threadIdx.x;
    long long step = (long long)gridDim.x * blockDim.x;
    for (; i < total; i += step) {
        int n = (int)(i % 288);
        long long q = i / 288;
        int t = (int)(q % 513);
        int b = (int)(q / 513);
        float v = 0.f;
        if (n < 257) {
            if (n == 256)      v = xv[t];
            else if (t == 512) v = tv[n];
            else               v = X2[((long long)b * 512 + t) * 256 + n];
        }
        Xf[i] = v; Xbf[i] = f2bf(v);
    }
}

// X += pos_enc(513,257) on the real region (pads stay zero)
__global__ void posenc_add_kernel(float* __restrict__ Xf,
                                  unsigned short* __restrict__ Xbf)
{
    long long total = 16ll * 513 * 288;
    long long i = (long long)blockIdx.x * blockDim.x + threadIdx.x;
    long long step = (long long)gridDim.x * blockDim.x;
    for (; i < total; i += step) {
        int n = (int)(i % 288);
        if (n >= 257) continue;
        int t = (int)((i / 288) % 513);
        float v = 0.f;
        if (n < 256) {
            int i2 = n & ~1;
            float freq = powf(10000.f, (2.f * (float)i2) / 257.f);
            float ang = (float)t / freq;
            v = (n & 1) ? cosf(ang) : sinf(ang);
        }
        float y = Xf[i] + v;
        Xf[i] = y; Xbf[i] = f2bf(y);
    }
}

// per row: center, emit bf16 xc (zero-padded to ld), emit L2 norm
// grid=(R, B), block=256
__global__ void center_rows_kernel(const float* __restrict__ X,
                                   unsigned short* __restrict__ xcbf,
                                   float* __restrict__ norm, int L, int ld)
{
    __shared__ float red[256];
    int r = blockIdx.x, b = blockIdx.y, tid = threadIdx.x;
    int R = gridDim.x;
    const float* row = X + ((long long)b * R + r) * ld;
    unsigned short* orow = xcbf + ((long long)b * R + r) * ld;
    float s = 0.f;
    for (int j = tid; j < L; j += 256) s += row[j];
    float mean = bsum(red, s) / (float)L;
    float ss = 0.f;
    for (int j = tid; j < L; j += 256) {
        float c = row[j] - mean;
        orow[j] = f2bf(c);
        ss += c * c;
    }
    for (int j = L + tid; j < ld; j += 256) orow[j] = 0;
    ss = bsum(red, ss);
    if (tid == 0) norm[(long long)b * R + r] = sqrtf(ss);
}

// Asm = softmax(relu(dot / (n_i*n_j))) per row.  grid=(R,B); dot/out ld = R
__global__ void corr_softmax_kernel(const float* __restrict__ dot,
                                    const float* __restrict__ norm,
                                    float* __restrict__ out, int R)
{
    __shared__ float buf[520];
    __shared__ float red[256];
    int i = blockIdx.x, b = blockIdx.y, tid = threadIdx.x;
    const float* drow = dot + ((long long)b * R + i) * R;
    float* orow = out + ((long long)b * R + i) * R;
    const float* nb = norm + (long long)b * R;
    float ni = nb[i];
    float mx = -3.4e38f;
    for (int j = tid; j < R; j += 256) {
        float v = fmaxf(drow[j] / (ni * nb[j] + 1e-30f), 0.f);
        buf[j] = v; mx = fmaxf(mx, v);
    }
    mx = bmax(red, mx);
    float sm = 0.f;
    for (int j = tid; j < R; j += 256) {
        float e = __expf(buf[j] - mx);
        buf[j] = e; sm += e;
    }
    sm = bsum(red, sm);
    float inv = 1.f / sm;
    for (int j = tid; j < R; j += 256) orow[j] = buf[j] * inv;
}

// attn = softmax(0.5*softmax(scores)+0.5*Asm) -> bf16 with ld Tp (pads zeroed)
// grid=(T, B); scores/Asm ld = T
__global__ void blend_softmax_kernel(const float* __restrict__ scores,
                                     const float* __restrict__ Asm,
                                     unsigned short* __restrict__ attn,
                                     int T, int Tp)
{
    __shared__ float buf[520];
    __shared__ float red[256];
    int t = blockIdx.x, b = blockIdx.y, tid = threadIdx.x;
    const float* srow = scores + ((long long)b * T + t) * T;
    const float* arow = Asm    + ((long long)b * T + t) * T;
    unsigned short* orow = attn + ((long long)b * T + t) * Tp;

    float mx = -3.4e38f;
    for (int j = tid; j < T; j += 256) mx = fmaxf(mx, srow[j]);
    mx = bmax(red, mx);
    float sm = 0.f;
    for (int j = tid; j < T; j += 256) {
        float e = __expf(srow[j] - mx);
        buf[j] = e; sm += e;
    }
    sm = bsum(red, sm);
    float inv = 1.f / sm;

    float mx2 = -3.4e38f;
    for (int j = tid; j < T; j += 256) {
        float v = 0.5f * buf[j] * inv + 0.5f * arow[j];
        buf[j] = v; mx2 = fmaxf(mx2, v);
    }
    mx2 = bmax(red, mx2);
    float sm2 = 0.f;
    for (int j = tid; j < T; j += 256) {
        float e = __expf(buf[j] - mx2);
        buf[j] = e; sm2 += e;
    }
    sm2 = bsum(red, sm2);
    float inv2 = 1.f / sm2;
    for (int j = tid; j < T; j += 256) orow[j] = f2bf(buf[j] * inv2);
    for (int j = T + tid; j < Tp; j += 256) orow[j] = 0;
}

// out = LayerNorm(x)*g + b + res ; writes f32 + bf16 (bf16 pads zeroed)
// grid=(T,B); all buffers ld-strided
__global__ void ln_residual_kernel(const float* __restrict__ X,
                                   const float* __restrict__ res,
                                   const float* __restrict__ g,
                                   const float* __restrict__ bta,
                                   float* __restrict__ outf,
                                   unsigned short* __restrict__ outbf,
                                   int L, int ld)
{
    __shared__ float red[256];
    int t = blockIdx.x, b = blockIdx.y, tid = threadIdx.x;
    int T = gridDim.x;
    long long base = ((long long)b * T + t) * ld;
    const float* row  = X   + base;
    const float* rrow = res + base;
    float s = 0.f;
    for (int j = tid; j < L; j += 256) s += row[j];
    float mean = bsum(red, s) / (float)L;
    float vv = 0.f;
    for (int j = tid; j < L; j += 256) {
        float c = row[j] - mean; vv += c * c;
    }
    float var = bsum(red, vv) / (float)L;
    float inv = rsqrtf(var + 1e-5f);
    for (int j = tid; j < L; j += 256) {
        float y = (row[j] - mean) * inv * g[j] + bta[j] + rrow[j];
        outf[base + j]  = y;
        outbf[base + j] = f2bf(y);
    }
    for (int j = L + tid; j < ld; j += 256) outbf[base + j] = 0;
}

// out[b] = sum_k hidden[b,k]*Wout2[k] + bout2  (hidden ld=128)
__global__ void final_dot_kernel(const unsigned short* __restrict__ h,
                                 const float* __restrict__ W2,
                                 const float* __restrict__ b2,
                                 float* __restrict__ out)
{
    __shared__ float red[128];
    int b = blockIdx.x, tid = threadIdx.x;
    float v = 0.f;
    if (tid < 100) v = bf2f(h[b * 128 + tid]) * W2[tid];
    red[tid] = v; __syncthreads();
    for (int s = 64; s > 0; s >>= 1) {
        if (tid < s) red[tid] += red[tid + s];
        __syncthreads();
    }
    if (tid == 0) out[b] = red[0] + b2[0];
}

// ---------------------------------------------------------------------------
// Host-side helpers
// ---------------------------------------------------------------------------
static const unsigned short* g_zguard = nullptr;   // set per launch (no state)

static void launch_gemm(hipStream_t s, const unsigned short* A,
                        const unsigned short* B, const float* bias,
                        float* C, unsigned short* Cbf,
                        int M, int N, int Kp, int lda, int ldb, int ldc, int ldcb,
                        long long sA, long long sB, long long sC, int batch,
                        int doGelu, int transC, int mzero, float alpha)
{
    dim3 g((N + 127) / 128, (M + 63) / 64, batch);
    gemm_bf16_wmma<<<g, 256, 0, s>>>(A, B, g_zguard, bias, C, Cbf, M, N, Kp,
                                     lda, ldb, ldc, ldcb, sA, sB, sC,
                                     doGelu, transC, mzero, alpha);
}

static void cast_bf(hipStream_t s, const float* in, unsigned short* out, long long n)
{
    cast_bf16_kernel<<<2048, 256, 0, s>>>(in, out, n);
}
static void cast_bT(hipStream_t s, const float* in, unsigned short* out,
                    int G, int Rin, int Rpad, int C)
{
    cast_bT_kernel<<<2048, 256, 0, s>>>(in, out, G, Rin, Rpad, C);
}

struct Scratch {
    unsigned short *Qbf, *Kbf, *Vt, *attnbf, *catbf, *ffbf;
    float *scores, *tmp;
};

// One VGT block (in place on Xf/Xbf; Yf/Ybf = ln1 output scratch)
static void run_vgt(hipStream_t s,
                    float* Xf, unsigned short* Xbf,
                    float* Yf, unsigned short* Ybf,
                    const float* Asm, int T, int Tp, int F, int Fp, int dff,
                    const unsigned short* WqT, const unsigned short* WkT,
                    const unsigned short* WvT,
                    const float* bq, const float* bk, const float* bv,
                    const unsigned short* WoT, const float* bo,
                    const float* ln1g, const float* ln1b,
                    const unsigned short* W1T, const float* b1,
                    const unsigned short* W2T, const float* b2,
                    const float* ln2g, const float* ln2b,
                    const Scratch& sc)
{
    const int Bn = 16, H = 8, D = 128, HD = 1024;
    const float isq = 1.f / sqrtf(128.f);
    for (int h = 0; h < H; ++h) {
        const long long wofs = (long long)h * D * Fp;   // per-head (128, Fp)
        // Q, K projections -> (T,128) bf16 ; V -> V^T (128, Tp) bf16, zero-pad
        launch_gemm(s, Xbf, WqT + wofs, bq + h * D, nullptr, sc.Qbf,
                    T, D, Fp, Fp, Fp, 0, D,
                    (long long)T * Fp, 0, (long long)T * D, Bn, 0, 0, T, 1.f);
        launch_gemm(s, Xbf, WkT + wofs, bk + h * D, nullptr, sc.Kbf,
                    T, D, Fp, Fp, Fp, 0, D,
                    (long long)T * Fp, 0, (long long)T * D, Bn, 0, 0, T, 1.f);
        launch_gemm(s, Xbf, WvT + wofs, bv + h * D, nullptr, sc.Vt,
                    T, D, Fp, Fp, Fp, 0, Tp,
                    (long long)T * Fp, 0, (long long)D * Tp, Bn, 0, 1, Tp, 1.f);
        // scores = Q @ K^T / sqrt(d)   (K already in (N,Kp) layout)
        launch_gemm(s, sc.Qbf, sc.Kbf, nullptr, sc.scores, nullptr,
                    T, T, D, D, D, T, 0,
                    (long long)T * D, (long long)T * D, (long long)T * T,
                    Bn, 0, 0, T, isq);
        blend_softmax_kernel<<<dim3(T, Bn), 256, 0, s>>>(sc.scores, Asm,
                                                         sc.attnbf, T, Tp);
        // hv = attn @ V, written straight into head-concat layout
        launch_gemm(s, sc.attnbf, sc.Vt, nullptr, nullptr, sc.catbf + h * D,
                    T, D, Tp, Tp, Tp, 0, HD,
                    (long long)T * Tp, (long long)D * Tp, (long long)T * HD,
                    Bn, 0, 0, T, 1.f);
    }
    // out-projection
    launch_gemm(s, sc.catbf, WoT, bo, sc.tmp, nullptr,
                T, F, HD, HD, HD, Fp, 0,
                (long long)T * HD, 0, (long long)T * Fp, Bn, 0, 0, T, 1.f);
    ln_residual_kernel<<<dim3(T, Bn), 256, 0, s>>>(sc.tmp, Xf, ln1g, ln1b,
                                                   Yf, Ybf, F, Fp);
    // FFN
    launch_gemm(s, Ybf, W1T, b1, nullptr, sc.ffbf,
                T, dff, Fp, Fp, Fp, 0, dff,
                (long long)T * Fp, 0, (long long)T * dff, Bn, 1, 0, T, 1.f);
    launch_gemm(s, sc.ffbf, W2T, b2, sc.tmp, nullptr,
                T, F, dff, dff, dff, Fp, 0,
                (long long)T * dff, 0, (long long)T * Fp, Bn, 0, 0, T, 1.f);
    ln_residual_kernel<<<dim3(T, Bn), 256, 0, s>>>(sc.tmp, Yf, ln2g, ln2b,
                                                   Xf, Xbf, F, Fp);
}

// ---------------------------------------------------------------------------
extern "C" void kernel_launch(void* const* d_in, const int* in_sizes, int n_in,
                              void* d_out, int out_size, void* d_ws, size_t ws_size,
                              hipStream_t stream)
{
    (void)in_sizes; (void)n_in; (void)out_size; (void)ws_size;
    const int B = 16, T1 = 513, N1 = 257, T1p = 544, N1p = 288;
    const int H = 8, D = 128, HD = 1024;

    // ---- inputs (setup_inputs dict order) ----
    const float* X     = (const float*)d_in[0];
    const float* Wt    = (const float*)d_in[1];
    const float* bt    = (const float*)d_in[2];
    const float* Wx    = (const float*)d_in[3];
    const float* bx    = (const float*)d_in[4];
    const float* t_v   = (const float*)d_in[5];
    const float* x_v   = (const float*)d_in[6];
    const float* Wout1 = (const float*)d_in[7];
    const float* bout1 = (const float*)d_in[8];
    const float* Wout2 = (const float*)d_in[9];
    const float* bout2 = (const float*)d_in[10];
    const float* tWq = (const float*)d_in[11]; const float* tbq = (const float*)d_in[12];
    const float* tWk = (const float*)d_in[13]; const float* tbk = (const float*)d_in[14];
    const float* tWv = (const float*)d_in[15]; const float* tbv = (const float*)d_in[16];
    const float* tWo = (const float*)d_in[17]; const float* tbo = (const float*)d_in[18];
    const float* tl1g = (const float*)d_in[19]; const float* tl1b = (const float*)d_in[20];
    const float* tl2g = (const float*)d_in[21]; const float* tl2b = (const float*)d_in[22];
    const float* tW1 = (const float*)d_in[23]; const float* tb1 = (const float*)d_in[24];
    const float* tW2 = (const float*)d_in[25]; const float* tb2 = (const float*)d_in[26];
    const float* sWq = (const float*)d_in[27]; const float* sbq = (const float*)d_in[28];
    const float* sWk = (const float*)d_in[29]; const float* sbk = (const float*)d_in[30];
    const float* sWv = (const float*)d_in[31]; const float* sbv = (const float*)d_in[32];
    const float* sWo = (const float*)d_in[33]; const float* sbo = (const float*)d_in[34];
    const float* sl1g = (const float*)d_in[35]; const float* sl1b = (const float*)d_in[36];
    const float* sl2g = (const float*)d_in[37]; const float* sl2b = (const float*)d_in[38];
    const float* sW1 = (const float*)d_in[39]; const float* sb1 = (const float*)d_in[40];
    const float* sW2 = (const float*)d_in[41]; const float* sb2 = (const float*)d_in[42];

    // ---- workspace bump allocator ----
    char* wsp = (char*)d_ws;
    size_t cur = 0;
    auto take = [&](size_t bytes) -> void* {
        void* r = wsp + cur;
        cur = (cur + bytes + 255) & ~(size_t)255;
        return r;
    };
    auto take_us = [&](long long n) { return (unsigned short*)take((size_t)n * 2); };
    auto take_f  = [&](long long n) { return (float*)take((size_t)n * 4); };

    // zero guard for branch-free OOB rows (>= max Kp + prefetch slack)
    const long long ZN = 513ll * 288 + 256;
    unsigned short* zguard = take_us(ZN);
    g_zguard = zguard;

    // bf16 operands: all B-side weights pre-transposed to (N, Kpad)
    unsigned short* Xin_bf = take_us(16ll * 256 * 512);
    unsigned short* WtT    = take_us(512ll * 512);
    unsigned short* WxT    = take_us(256ll * 256);
    unsigned short* Wout1T = take_us(100ll * 513 * 288);
    unsigned short* tWqT = take_us(16ll * 128 * 288);
    unsigned short* tWkT = take_us(16ll * 128 * 288);
    unsigned short* tWvT = take_us(16ll * 128 * 288);
    unsigned short* tWoT = take_us(2ll * 257 * 1024);
    unsigned short* tW1T = take_us(2ll * 1024 * 288);
    unsigned short* tW2T = take_us(2ll * 257 * 1024);
    unsigned short* sWqT = take_us(16ll * 128 * 544);
    unsigned short* sWkT = take_us(16ll * 128 * 544);
    unsigned short* sWvT = take_us(16ll * 128 * 544);
    unsigned short* sWoT = take_us(2ll * 513 * 1024);
    unsigned short* sW1T = take_us(2ll * 2048 * 544);
    unsigned short* sW2T = take_us(2ll * 513 * 2048);

    // activations
    float* X0 = take_f(16ll * 256 * 512);                 // reused as X2
    float* X1 = take_f(16ll * 512 * 256);
    unsigned short* X1bf = take_us(16ll * 512 * 256);
    long long nT = (long long)B * T1 * N1p;               // (513, 288) per sample
    long long nS = (long long)B * N1 * T1p;               // (257, 544) per sample
    float* XA = take_f(nT);  unsigned short* XAbf = take_us(nT);
    float* XB = take_f(nT);  unsigned short* XBbf = take_us(nT);
    float* XSA = take_f(nS); unsigned short* XSAbf = take_us(nS);
    float* XSB = take_f(nS); unsigned short* XSBbf = take_us(nS);
    unsigned short* xcT = take_us(nT);
    unsigned short* xcS = take_us(nS);
    float* normT = take_f((long long)B * T1);
    float* normS = take_f((long long)B * N1);
    float* AsmT = take_f((long long)B * T1 * T1);
    float* AsmS = take_f((long long)B * N1 * N1);
    float* big0 = take_f((long long)B * T1 * T1);          // dot / scores
    unsigned short* attnbf = take_us((long long)B * T1 * T1p);
    unsigned short* Qbf = take_us((long long)B * T1 * D);
    unsigned short* Kbf = take_us((long long)B * T1 * D);
    unsigned short* Vt  = take_us((long long)B * D * T1p); // V^T (128, Tp)
    unsigned short* catbf = take_us((long long)B * T1 * HD);
    unsigned short* ffbf  = take_us((long long)B * 257 * 2048); // >= 16*513*1024
    float* tmp = take_f(nT);                                // >= (257,544) too
    unsigned short* hbf = take_us(16ll * 128);

    // ---- init guard + casts (weights transposed + zero-padded) ----
    zero_us_kernel<<<512, 256, 0, stream>>>(zguard, ZN);
    cast_bf(stream, X, Xin_bf, 16ll * 256 * 512);
    cast_bT(stream, Wt, WtT, 1, 512, 512, 512);
    cast_bT(stream, Wx, WxT, 1, 256, 256, 256);
    cast_wout1_kernel<<<2048, 256, 0, stream>>>(Wout1, Wout1T);
    cast_bT(stream, tWq, tWqT, 16, 257, 288, 128);
    cast_bT(stream, tWk, tWkT, 16, 257, 288, 128);
    cast_bT(stream, tWv, tWvT, 16, 257, 288, 128);
    cast_bT(stream, tWo, tWoT, 2, 1024, 1024, 257);
    cast_bT(stream, tW1, tW1T, 2, 257, 288, 1024);
    cast_bT(stream, tW2, tW2T, 2, 1024, 1024, 257);
    cast_bT(stream, sWq, sWqT, 16, 513, 544, 128);
    cast_bT(stream, sWk, sWkT, 16, 513, 544, 128);
    cast_bT(stream, sWv, sWvT, 16, 513, 544, 128);
    cast_bT(stream, sWo, sWoT, 2, 1024, 1024, 513);
    cast_bT(stream, sW1, sW1T, 2, 513, 544, 2048);
    cast_bT(stream, sW2, sW2T, 2, 2048, 2048, 513);

    // ---- stem ----
    launch_gemm(stream, Xin_bf, WtT, bt, X0, nullptr,
                256, 512, 512, 512, 512, 512, 0,
                256ll * 512, 0, 256ll * 512, B, 0, 0, 256, 1.f);
    {
        dim3 g(16, 8, B);
        transpose_kernel<<<g, 256, 0, stream>>>(X0, X1, X1bf, 256, 512, 512, 256);
    }
    launch_gemm(stream, X1bf, WxT, bx, X0, nullptr,
                512, 256, 256, 256, 256, 256, 0,
                512ll * 256, 0, 512ll * 256, B, 0, 0, 512, 1.f);
    build_xc_kernel<<<2048, 256, 0, stream>>>(X0, t_v, x_v, XA, XAbf);

    // ---- Pearson correlation adjacencies (pre pos-enc) ----
    center_rows_kernel<<<dim3(T1, B), 256, 0, stream>>>(XA, xcT, normT, N1, N1p);
    launch_gemm(stream, xcT, xcT, nullptr, big0, nullptr,
                T1, T1, N1p, N1p, N1p, T1, 0,
                (long long)T1 * N1p, (long long)T1 * N1p, (long long)T1 * T1,
                B, 0, 0, T1, 1.f);
    corr_softmax_kernel<<<dim3(T1, B), 256, 0, stream>>>(big0, normT, AsmT, T1);

    {
        dim3 g((N1 + 31) / 32, T1p / 32, B);
        transpose_kernel<<<g, 256, 0, stream>>>(XA, XSA, XSAbf, T1, N1, N1p, T1p);
    }
    center_rows_kernel<<<dim3(N1, B), 256, 0, stream>>>(XSA, xcS, normS, T1, T1p);
    launch_gemm(stream, xcS, xcS, nullptr, big0, nullptr,
                N1, N1, T1p, T1p, T1p, N1, 0,
                (long long)N1 * T1p, (long long)N1 * T1p, (long long)N1 * N1,
                B, 0, 0, N1, 1.f);
    corr_softmax_kernel<<<dim3(N1, B), 256, 0, stream>>>(big0, normS, AsmS, N1);

    posenc_add_kernel<<<2048, 256, 0, stream>>>(XA, XAbf);

    // ---- transformer blocks ----
    Scratch sc{Qbf, Kbf, Vt, attnbf, catbf, ffbf, big0, tmp};
    for (int i = 0; i < 2; ++i) {
        // temporal block: rows T1, features N1, dff 1024
        run_vgt(stream, XA, XAbf, XB, XBbf, AsmT, T1, T1p, N1, N1p, 1024,
                tWqT + (long long)i * H * D * N1p,
                tWkT + (long long)i * H * D * N1p,
                tWvT + (long long)i * H * D * N1p,
                tbq + (long long)i * H * D, tbk + (long long)i * H * D,
                tbv + (long long)i * H * D,
                tWoT + (long long)i * N1 * HD, tbo + i * N1,
                tl1g + i * N1, tl1b + i * N1,
                tW1T + (long long)i * 1024 * N1p, tb1 + i * 1024,
                tW2T + (long long)i * N1 * 1024, tb2 + i * N1,
                tl2g + i * N1, tl2b + i * N1, sc);
        {
            dim3 g((N1 + 31) / 32, T1p / 32, B);
            transpose_kernel<<<g, 256, 0, stream>>>(XA, XSA, XSAbf, T1, N1, N1p, T1p);
        }
        // spatial block: rows N1, features T1, dff 2048
        run_vgt(stream, XSA, XSAbf, XSB, XSBbf, AsmS, N1, N1p, T1, T1p, 2048,
                sWqT + (long long)i * H * D * T1p,
                sWkT + (long long)i * H * D * T1p,
                sWvT + (long long)i * H * D * T1p,
                sbq + (long long)i * H * D, sbk + (long long)i * H * D,
                sbv + (long long)i * H * D,
                sWoT + (long long)i * T1 * HD, sbo + i * T1,
                sl1g + i * T1, sl1b + i * T1,
                sW1T + (long long)i * 2048 * T1p, sb1 + i * 2048,
                sW2T + (long long)i * T1 * 2048, sb2 + i * T1,
                sl2g + i * T1, sl2b + i * T1, sc);
        {
            dim3 g((T1 + 31) / 32, N1p / 32, B);
            transpose_kernel<<<g, 256, 0, stream>>>(XSA, XA, XAbf, N1, T1, T1p, N1p);
        }
    }

    // ---- head: (16, 513*288) @ Wout1T^T -> gelu -> dot with Wout2 ----
    const int KH = 513 * 288;   // 147744, multiple of 32, zero-padded cols
    launch_gemm(stream, XAbf, Wout1T, bout1, nullptr, hbf,
                16, 100, KH, KH, KH, 0, 128,
                0, 0, 0, 1, 1, 0, 16, 1.f);
    final_dot_kernel<<<16, 128, 0, stream>>>(hbf, Wout2, bout2, (float*)d_out);
}
